// TwoLayerSNN_70712341561965
// MI455X (gfx1250) — matrix operand order repaired and checked by reading, not compiled, not run on hardware
//
#include <hip/hip_runtime.h>

// ---------------------------------------------------------------------------
// Two-layer SNN forward on gfx1250 (MI455X), bf16 WMMA (v_wmma_f32_16x16x32_bf16)
// Shapes: B=256, T=128, D_in=1024, H0=H1=2048, D_out=512
// ---------------------------------------------------------------------------

typedef __bf16 bf16;
typedef bf16  v16bf __attribute__((ext_vector_type(16)));
typedef bf16  v8bf  __attribute__((ext_vector_type(8)));
typedef float v8f   __attribute__((ext_vector_type(8)));

union ABfrag { v16bf v; v8bf h[2]; };

#define SNN_B    256
#define SNN_T    128
#define SNN_DIN  1024
#define SNN_H    2048
#define SNN_DOUT 512
#define SNN_BETA 0.9f

// ---- WMMA wrapper -----------------------------------------------------------
__device__ inline v8f wmma_bf16(v16bf a, v16bf b, v8f c) {
  // 8 args: (neg_a, A, neg_b, B, c_mod, C, reuse_a, reuse_b)
  return __builtin_amdgcn_wmma_f32_16x16x32_bf16(
      false, a, false, b, (short)0, c, false, false);
}

// ---- A fragment: 16x32 (MxK) bf16, row-major source, leading dim lda --------
// lane<16:  kb=k0:   holds K = kb..kb+7 and kb+16..kb+23  (ISA 16-bit A layout)
// lane>=16: kb=k0+8
__device__ inline v16bf load_a_bf16(const bf16* __restrict__ src, int lda,
                                    int m0, int k0, int lane) {
  int row = m0 + (lane & 15);
  int kb  = k0 + ((lane >> 4) << 3);
  const bf16* p = src + row * lda + kb;
  ABfrag f;
  f.h[0] = *reinterpret_cast<const v8bf*>(p);        // K = kb .. kb+7
  f.h[1] = *reinterpret_cast<const v8bf*>(p + 16);   // K = kb+16 .. kb+23
  return f.v;
}

// A fragment converted from f32 on the fly (no scale — fallback x path)
__device__ inline v16bf load_a_f32(const float* __restrict__ src, int lda,
                                   int m0, int k0, int lane) {
  int row = m0 + (lane & 15);
  int kb  = k0 + ((lane >> 4) << 3);
  const float* p = src + row * lda + kb;
  ABfrag f;
#pragma unroll
  for (int i = 0; i < 8; ++i) f.v[i]     = (bf16)p[i];
#pragma unroll
  for (int i = 0; i < 8; ++i) f.v[8 + i] = (bf16)p[16 + i];
  return f.v;
}

// A fragment converted from f32 with scale (final FC: exact x 1/T)
__device__ inline v16bf load_a_f32s(const float* __restrict__ src, int lda,
                                    int m0, int k0, int lane, float scale) {
  int row = m0 + (lane & 15);
  int kb  = k0 + ((lane >> 4) << 3);
  const float* p = src + row * lda + kb;
  ABfrag f;
#pragma unroll
  for (int i = 0; i < 8; ++i) f.v[i]     = (bf16)(p[i]      * scale);
#pragma unroll
  for (int i = 0; i < 8; ++i) f.v[8 + i] = (bf16)(p[16 + i] * scale);
  return f.v;
}

// ---- B fragment: 32x16 (KxN) where B[k][n] = W[n][k], W row-major [N, K] ----
// lane<16: N=lane,    K = k0..k0+7 and k0+16..k0+23
// lane>=16:N=lane-16, K = k0+8..k0+15 and k0+24..k0+31
__device__ inline v16bf load_b(const bf16* __restrict__ W, int ldw,
                               int n0, int k0, int lane) {
  int n  = n0 + (lane & 15);
  int kb = k0 + ((lane >> 4) << 3);
  const bf16* p = W + n * ldw + kb;
  ABfrag f;
  f.h[0] = *reinterpret_cast<const v8bf*>(p);
  f.h[1] = *reinterpret_cast<const v8bf*>(p + 16);
  return f.v;
}

// ---------------------------------------------------------------------------
// Fused layer kernel: C = z_prev @ Wz^T + A2 @ Wx^T ; then LIF update:
//   v = BETA*v + C + (bz+bx) - z_prev ; z_out = (v-1 > 0) ; acc += z_out
// Each wave computes a 16x64 strip of the 256 x H output.
// ---------------------------------------------------------------------------
template <bool A2F32>
__global__ __launch_bounds__(256) void snn_layer_kernel(
    const bf16* __restrict__ Wz, int Kz,
    const bf16* __restrict__ Wx, int Kx,
    const void* __restrict__ a2, int ldA2,   // f32 x_t or bf16 (row-major)
    const float* __restrict__ bz,
    const float* __restrict__ bx,
    const bf16* __restrict__ z_prev,         // [256, Kz] (Kz == H)
    float* __restrict__ v,                   // [256, H]
    bf16* __restrict__ z_out,                // [256, H]
    float* __restrict__ acc,                 // nullable
    int H) {
  const int lane    = threadIdx.x & 31;
  const int wid     = threadIdx.x >> 5;
  const int nstrips = H >> 6;
  int task  = blockIdx.x * 8 + wid;
  int mtile = task / nstrips;
  int nstrp = task % nstrips;
  if (mtile >= (SNN_B >> 4)) return;
  const int m0 = mtile << 4;
  const int n0 = nstrp << 6;

  v8f c[4] = {v8f{}, v8f{}, v8f{}, v8f{}};

  // --- recurrent part: z_prev @ Wz^T -------------------------------------
  for (int k0 = 0; k0 < Kz; k0 += 32) {
    v16bf a = load_a_bf16(z_prev, Kz, m0, k0, lane);
#pragma unroll
    for (int nt = 0; nt < 4; ++nt) {
      v16bf b = load_b(Wz, Kz, n0 + nt * 16, k0, lane);
      c[nt] = wmma_bf16(a, b, c[nt]);
    }
  }
  // --- feed-forward part: A2 @ Wx^T --------------------------------------
  for (int k0 = 0; k0 < Kx; k0 += 32) {
    v16bf a;
    if (A2F32) a = load_a_f32((const float*)a2, ldA2, m0, k0, lane);
    else       a = load_a_bf16((const bf16*)a2, ldA2, m0, k0, lane);
#pragma unroll
    for (int nt = 0; nt < 4; ++nt) {
      v16bf b = load_b(Wx, Kx, n0 + nt * 16, k0, lane);
      c[nt] = wmma_bf16(a, b, c[nt]);
    }
  }

  // --- LIF epilogue; C/D layout: c[r] <-> (M = r + 8*(lane>>4), N = lane&15)
  const int hi = (lane >> 4) << 3;
  const int nc = lane & 15;
#pragma unroll
  for (int nt = 0; nt < 4; ++nt) {
    int n = n0 + nt * 16 + nc;
    float bias = bz[n] + bx[n];
#pragma unroll
    for (int r = 0; r < 8; ++r) {
      int m   = m0 + hi + r;
      int idx = m * H + n;
      float zp = (float)z_prev[idx];                 // THRESHOLD == 1
      float vn = SNN_BETA * v[idx] + c[nt][r] + bias - zp;
      v[idx] = vn;
      float s = (vn - 1.0f) > 0.0f ? 1.0f : 0.0f;
      z_out[idx] = (bf16)s;
      if (acc) acc[idx] += s;
    }
  }
}

// ---------------------------------------------------------------------------
// Final FC: out = (acc * scale) @ Wfc^T + bfc       (acc/128 is exact in bf16)
// ---------------------------------------------------------------------------
__global__ __launch_bounds__(256) void snn_fc_kernel(
    const float* __restrict__ accf,   // [256, K]
    const bf16*  __restrict__ Wfc,    // [N, K]
    const float* __restrict__ bfc,
    float*       __restrict__ out,    // [256, N]
    int K, int N, float scale) {
  const int lane    = threadIdx.x & 31;
  const int wid     = threadIdx.x >> 5;
  const int nstrips = N >> 6;
  int task  = blockIdx.x * 8 + wid;
  int mtile = task / nstrips;
  int nstrp = task % nstrips;
  if (mtile >= (SNN_B >> 4)) return;
  const int m0 = mtile << 4;
  const int n0 = nstrp << 6;

  v8f c[4] = {v8f{}, v8f{}, v8f{}, v8f{}};
  for (int k0 = 0; k0 < K; k0 += 32) {
    v16bf a = load_a_f32s(accf, K, m0, k0, lane, scale);
#pragma unroll
    for (int nt = 0; nt < 4; ++nt) {
      v16bf b = load_b(Wfc, K, n0 + nt * 16, k0, lane);
      c[nt] = wmma_bf16(a, b, c[nt]);
    }
  }
  const int hi = (lane >> 4) << 3;
  const int nc = lane & 15;
#pragma unroll
  for (int nt = 0; nt < 4; ++nt) {
    int n = n0 + nt * 16 + nc;
    float bias = bfc[n];
#pragma unroll
    for (int r = 0; r < 8; ++r)
      out[(m0 + hi + r) * N + n] = c[nt][r] + bias;
  }
}

// ---- utility kernels --------------------------------------------------------
__global__ void cvt_f32_bf16_kernel(const float* __restrict__ src,
                                    bf16* __restrict__ dst, int n) {
  for (int i = blockIdx.x * blockDim.x + threadIdx.x; i < n;
       i += gridDim.x * blockDim.x)
    dst[i] = (bf16)src[i];
}

// x [B, T, D] f32  ->  xb [T, B, D] bf16   (write-coalesced)
__global__ void cvt_x_kernel(const float* __restrict__ x,
                             bf16* __restrict__ xb, int total) {
  for (int i = blockIdx.x * blockDim.x + threadIdx.x; i < total;
       i += gridDim.x * blockDim.x) {
    int k    = i & (SNN_DIN - 1);
    int rest = i >> 10;                 // DIN = 1024
    int m    = rest & (SNN_B - 1);
    int t    = rest >> 8;               // B = 256
    xb[i] = (bf16)x[((size_t)m * SNN_T + t) * SNN_DIN + k];
  }
}

__global__ void zero_u32_kernel(unsigned* __restrict__ p, int n) {
  for (int i = blockIdx.x * blockDim.x + threadIdx.x; i < n;
       i += gridDim.x * blockDim.x)
    p[i] = 0u;
}

// ---------------------------------------------------------------------------
extern "C" void kernel_launch(void* const* d_in, const int* in_sizes, int n_in,
                              void* d_out, int out_size, void* d_ws,
                              size_t ws_size, hipStream_t stream) {
  (void)in_sizes; (void)n_in; (void)out_size;

  const float* x   = (const float*)d_in[0];   // [B, T, D_in]
  const float* Wx1 = (const float*)d_in[1];   // [H0, D_in]
  const float* bx1 = (const float*)d_in[2];
  const float* Wz1 = (const float*)d_in[3];   // [H0, H0]
  const float* bz1 = (const float*)d_in[4];
  const float* Wx2 = (const float*)d_in[5];   // [H1, H0]
  const float* bx2 = (const float*)d_in[6];
  const float* Wz2 = (const float*)d_in[7];   // [H1, H1]
  const float* bz2 = (const float*)d_in[8];
  const float* Wfc = (const float*)d_in[9];   // [D_out, H1]
  const float* bfc = (const float*)d_in[10];
  float* out = (float*)d_out;                 // [B, D_out]

  // ---- workspace layout (all sizes 256B-multiples) ------------------------
  char* ws = (char*)d_ws;
  size_t o = 0;
  bf16* Wz1b = (bf16*)(ws + o); o += (size_t)SNN_H * SNN_H   * 2;  // 8 MB
  bf16* Wx1b = (bf16*)(ws + o); o += (size_t)SNN_H * SNN_DIN * 2;  // 4 MB
  bf16* Wz2b = (bf16*)(ws + o); o += (size_t)SNN_H * SNN_H   * 2;  // 8 MB
  bf16* Wx2b = (bf16*)(ws + o); o += (size_t)SNN_H * SNN_H   * 2;  // 8 MB
  bf16* Wfcb = (bf16*)(ws + o); o += (size_t)SNN_DOUT * SNN_H * 2; // 2 MB
  char* state0 = ws + o;                                           // zeroed
  float* v0  = (float*)(ws + o); o += (size_t)SNN_B * SNN_H * 4;   // 2 MB
  float* v1  = (float*)(ws + o); o += (size_t)SNN_B * SNN_H * 4;   // 2 MB
  float* acc = (float*)(ws + o); o += (size_t)SNN_B * SNN_H * 4;   // 2 MB
  bf16* z0buf[2]; bf16* z1buf[2];
  z0buf[0] = (bf16*)(ws + o); o += (size_t)SNN_B * SNN_H * 2;      // 1 MB
  z0buf[1] = (bf16*)(ws + o); o += (size_t)SNN_B * SNN_H * 2;
  z1buf[0] = (bf16*)(ws + o); o += (size_t)SNN_B * SNN_H * 2;
  z1buf[1] = (bf16*)(ws + o); o += (size_t)SNN_B * SNN_H * 2;
  size_t state_bytes = (size_t)(ws + o - state0);

  // Optional pre-converted x [T, B, D_in] bf16 (67 MB) — only if ws allows.
  bf16* xb = (bf16*)(ws + o);
  size_t need_with_xb = o + (size_t)SNN_T * SNN_B * SNN_DIN * 2;
  const bool use_xb = (ws_size >= need_with_xb);

  // ---- one-time prep: convert weights (and maybe x) to bf16, zero state ----
  auto cvt = [&](const float* s, bf16* d, int n) {
    cvt_f32_bf16_kernel<<<2048, 256, 0, stream>>>(s, d, n);
  };
  cvt(Wz1, Wz1b, SNN_H * SNN_H);
  cvt(Wx1, Wx1b, SNN_H * SNN_DIN);
  cvt(Wz2, Wz2b, SNN_H * SNN_H);
  cvt(Wx2, Wx2b, SNN_H * SNN_H);
  cvt(Wfc, Wfcb, SNN_DOUT * SNN_H);
  if (use_xb)
    cvt_x_kernel<<<4096, 256, 0, stream>>>(x, xb, SNN_T * SNN_B * SNN_DIN);
  zero_u32_kernel<<<2048, 256, 0, stream>>>((unsigned*)state0,
                                            (int)(state_bytes / 4));

  // ---- time scan: 2 fused GEMM+LIF kernels per step ------------------------
  // grid: 16 M-tiles x (H/64) N-strips, 8 waves (16x64 strips) per block
  const int layer_blocks = (16 * (SNN_H / 64)) / 8;  // 64
  for (int t = 0; t < SNN_T; ++t) {
    int cur = t & 1, nxt = cur ^ 1;
    // layer 1: A2 = x_t, recurrent z0
    if (use_xb) {
      const bf16* xt = xb + (size_t)t * SNN_B * SNN_DIN;   // [B, D_in] bf16
      snn_layer_kernel<false><<<layer_blocks, 256, 0, stream>>>(
          Wz1b, SNN_H, Wx1b, SNN_DIN, (const void*)xt, SNN_DIN,
          bz1, bx1, z0buf[cur], v0, z0buf[nxt], (float*)nullptr, SNN_H);
    } else {
      const float* xt = x + (size_t)t * SNN_DIN;           // row stride T*D_in
      snn_layer_kernel<true><<<layer_blocks, 256, 0, stream>>>(
          Wz1b, SNN_H, Wx1b, SNN_DIN, (const void*)xt, SNN_T * SNN_DIN,
          bz1, bx1, z0buf[cur], v0, z0buf[nxt], (float*)nullptr, SNN_H);
    }
    // layer 2: A2 = z0_new (bf16), recurrent z1, accumulate spikes
    snn_layer_kernel<false><<<layer_blocks, 256, 0, stream>>>(
        Wz2b, SNN_H, Wx2b, SNN_H, (const void*)z0buf[nxt], SNN_H,
        bz2, bx2, z1buf[cur], v1, z1buf[nxt], acc, SNN_H);
  }

  // ---- final FC: out = (acc/T) @ Wfc^T + bfc -------------------------------
  const int fc_blocks = (16 * (SNN_DOUT / 64)) / 8;  // 16
  snn_fc_kernel<<<fc_blocks, 256, 0, stream>>>(
      acc, Wfcb, bfc, out, SNN_H, SNN_DOUT, 1.0f / (float)SNN_T);
}
